// MoeLayer_19095424598752
// MI455X (gfx1250) — matrix operand order repaired and checked by reading, not compiled, not run on hardware
//
#include <hip/hip_runtime.h>
#include <hip/hip_bf16.h>
#include <math.h>

// ---------- problem constants (from reference) ----------
constexpr int Bsz = 8, S = 2048, D = 1024, E = 8, F = 4096;
constexpr int T = Bsz * S;          // 16384 tokens
constexpr int CAP = T / E;          // 2048 capacity per expert

// ---------- vector types ----------
typedef __attribute__((ext_vector_type(16))) __bf16 v16bf;
typedef __attribute__((ext_vector_type(8)))  __bf16 v8bf;
typedef __attribute__((ext_vector_type(4)))  __bf16 v4bf;
typedef __attribute__((ext_vector_type(8)))  float  v8f;
typedef __attribute__((ext_vector_type(4)))  int    v4i;

// ---------- CDNA5 async LDS copy (guarded; falls back to sync copy) ----------
#if __has_builtin(__builtin_amdgcn_global_load_async_to_lds_b128)
#define USE_ASYNC_COPY 1
#else
#define USE_ASYNC_COPY 0
#endif

__device__ __forceinline__ void stage16(const __bf16* g, __bf16* l) {
#if USE_ASYNC_COPY
  typedef __attribute__((address_space(1))) v4i gv4i_t;   // global <4 x i32>
  typedef __attribute__((address_space(3))) v4i lv4i_t;   // LDS    <4 x i32>
  __builtin_amdgcn_global_load_async_to_lds_b128(
      (gv4i_t*)(uintptr_t)g,
      (lv4i_t*)(unsigned int)(uintptr_t)l,
      0, 0);
#else
  *(v8bf*)l = *(const v8bf*)g;
#endif
}

__device__ __forceinline__ void async_wait_all() {
#if USE_ASYNC_COPY
#if __has_builtin(__builtin_amdgcn_s_wait_asynccnt)
  __builtin_amdgcn_s_wait_asynccnt(0);
#else
  asm volatile("s_wait_asynccnt 0" ::: "memory");
#endif
#endif
}

__device__ __forceinline__ float gelu_tanh(float x) {
  const float c0 = 0.7978845608028654f;  // sqrt(2/pi)
  const float c1 = 0.044715f;
  return 0.5f * x * (1.0f + tanhf(c0 * (x + c1 * x * x * x)));
}

// ---------- 1) zero the output ----------
__global__ void zero_f32_kernel(float4* __restrict__ p, int n4) {
  int i = blockIdx.x * blockDim.x + threadIdx.x;
  int stride = gridDim.x * blockDim.x;
  float4 z; z.x = z.y = z.z = z.w = 0.0f;
  for (; i < n4; i += stride) p[i] = z;
}

// ---------- 2) router: one wave per token ----------
__global__ __launch_bounds__(256)
void router_kernel(const float* __restrict__ x, const float* __restrict__ rw,
                   int* __restrict__ eidx, float* __restrict__ gate) {
  const int wave = threadIdx.x >> 5;
  const int lane = threadIdx.x & 31;
  const int t = blockIdx.x * 8 + wave;
  const float* xr = x + (size_t)t * D;
  float acc[E];
#pragma unroll
  for (int e = 0; e < E; ++e) acc[e] = 0.0f;
  for (int d = lane; d < D; d += 32) {
    float xv = xr[d];
    const float* rwr = rw + d * E;
#pragma unroll
    for (int e = 0; e < E; ++e) acc[e] += xv * rwr[e];
  }
#pragma unroll
  for (int off = 16; off > 0; off >>= 1) {
#pragma unroll
    for (int e = 0; e < E; ++e) acc[e] += __shfl_xor(acc[e], off, 32);
  }
  if (lane == 0) {
    float lmax = acc[0]; int bi = 0;
#pragma unroll
    for (int e = 1; e < E; ++e) { if (acc[e] > lmax) { lmax = acc[e]; bi = e; } }
    float s = 0.0f;
#pragma unroll
    for (int e = 0; e < E; ++e) s += expf(acc[e] - lmax);
    eidx[t] = bi;
    gate[t] = 1.0f / s;   // softmax prob of the argmax
  }
}

// ---------- 3) capacity scan: one thread per expert, sequential (order matters) ----------
__global__ void positions_kernel(const int* __restrict__ eidx, const float* __restrict__ gate,
                                 int* __restrict__ s2t, float* __restrict__ sgate) {
  int e = threadIdx.x;
  if (e >= E) return;
  int cnt = 0;
  for (int t = 0; t < T; ++t) {
    if (eidx[t] == e) {
      if (cnt < CAP) { s2t[e * CAP + cnt] = t; sgate[e * CAP + cnt] = gate[t]; }
      ++cnt;
    }
  }
  for (int p = (cnt < CAP ? cnt : CAP); p < CAP; ++p) {
    s2t[e * CAP + p] = -1; sgate[e * CAP + p] = 0.0f;
  }
}

// ---------- 4a) dispatch: gather token rows into bf16 [E*CAP, D] ----------
__global__ __launch_bounds__(256)
void dispatch_kernel(const float* __restrict__ x, const int* __restrict__ s2t,
                     __bf16* __restrict__ Xe) {
  const int slot = blockIdx.x;
  const int t = s2t[slot];
  const int i = threadIdx.x * 4;
  v4bf o;
  if (t >= 0) {
    const float4 v = *(const float4*)(x + (size_t)t * D + i);
    o[0] = (__bf16)v.x; o[1] = (__bf16)v.y; o[2] = (__bf16)v.z; o[3] = (__bf16)v.w;
  } else {
    o[0] = o[1] = o[2] = o[3] = (__bf16)0.0f;
  }
  *(v4bf*)(Xe + (size_t)slot * D + i) = o;
}

// ---------- 4b) fp32 -> bf16 weight conversion ----------
__global__ void cvt_bf16_kernel(const float4* __restrict__ src, v4bf* __restrict__ dst, int n4) {
  int i = blockIdx.x * blockDim.x + threadIdx.x;
  int stride = gridDim.x * blockDim.x;
  for (; i < n4; i += stride) {
    float4 v = src[i];
    v4bf o;
    o[0] = (__bf16)v.x; o[1] = (__bf16)v.y; o[2] = (__bf16)v.z; o[3] = (__bf16)v.w;
    dst[i] = o;
  }
}

// ---------- 5) grouped GEMM: 128x128x32 tiles, 8 waves, double-buffered LDS ----------
// MODE 0: hidden = gelu(Xe @ W1)  -> bf16 [E, CAP, F]
// MODE 1: y[t]   = (hidden @ W2)[e, c, :] * sgate  (scatter to tokens)
constexpr int BM = 128, BN = 128, BK = 32;
constexpr int SB_STRIDE = 40;  // bf16 elems per column of sB (even, 16B-aligned rows)

template <int MODE>
__global__ __launch_bounds__(256)
void moe_gemm_kernel(const __bf16* __restrict__ Aall, const __bf16* __restrict__ Ball,
                     size_t Aes, size_t Bes, int K, int lda, int ldb,
                     __bf16* __restrict__ hidden,
                     float* __restrict__ out,
                     const int* __restrict__ s2t,
                     const float* __restrict__ sgate) {
  __shared__ __attribute__((aligned(16))) __bf16 sA[2][BM * BK];        // row-major [m][k]
  __shared__ __attribute__((aligned(16))) __bf16 sB[2][BN * SB_STRIDE]; // [n][k], K-contig

  const int e  = blockIdx.z;
  const int m0 = blockIdx.y * BM;
  const int n0 = blockIdx.x * BN;
  const __bf16* A = Aall + (size_t)e * Aes;
  const __bf16* B = Ball + (size_t)e * Bes;

  const int tid  = threadIdx.x;
  const int wave = tid >> 5;
  const int lane = tid & 31;
  const int wm = wave >> 2;      // 0..1 -> 64 rows each
  const int wn = wave & 3;       // 0..3 -> 32 cols each
  const int lh = lane & 15;      // lane within half
  const int hh = lane >> 4;      // which half of the wave

  v8f acc[4][2];
#pragma unroll
  for (int i = 0; i < 4; ++i)
#pragma unroll
    for (int j = 0; j < 2; ++j) acc[i][j] = (v8f){};

  // stage one K-tile (A via async-to-LDS, B transposed with packed b32 stores)
  auto stage_tiles = [&](int kk, int buf) {
    // A tile: 128x32 bf16, straight copy, 16B per lane per iter
#pragma unroll
    for (int it = 0; it < 2; ++it) {
      int idx = tid + it * 256;
      int r = idx >> 2;
      int q = (idx & 3) * 8;
      stage16(A + (size_t)(m0 + r) * lda + kk + q, &sA[buf][r * BK + q]);
    }
    // B tile: 32x128 bf16 -> sB[n][k]; two K-rows per thread, pack K-pairs into b32
    int r2 = tid >> 4;             // k-pair index 0..15  (k = 2*r2)
    int c  = (tid & 15) * 8;       // n col 0..120
    const __bf16* bp = B + (size_t)(kk + 2 * r2) * ldb + n0 + c;
    v8bf lo = *(const v8bf*)bp;
    v8bf hi = *(const v8bf*)(bp + ldb);
    const unsigned short* ls = (const unsigned short*)&lo;
    const unsigned short* hs = (const unsigned short*)&hi;
    unsigned* sB32 = (unsigned*)&sB[buf][0];
#pragma unroll
    for (int j = 0; j < 8; ++j)
      sB32[((c + j) * SB_STRIDE + 2 * r2) >> 1] =
          (unsigned)ls[j] | ((unsigned)hs[j] << 16);
  };

  const int nk = K / BK;
  stage_tiles(0, 0);
  async_wait_all();
  __syncthreads();

  for (int ki = 0; ki < nk; ++ki) {
    const int cur = ki & 1;
    if (ki + 1 < nk) stage_tiles((ki + 1) * BK, cur ^ 1);

    // --- B fragments: element i <-> K = hh*16 + i, N = lh ---
    v16bf bfrag[2];
#pragma unroll
    for (int nt = 0; nt < 2; ++nt) {
      int n = wn * 32 + nt * 16 + lh;
      v8bf blo = *(const v8bf*)&sB[cur][n * SB_STRIDE + hh * 16];
      v8bf bhi = *(const v8bf*)&sB[cur][n * SB_STRIDE + hh * 16 + 8];
      bfrag[nt] = __builtin_shufflevector(blo, bhi, 0, 1, 2, 3, 4, 5, 6, 7,
                                          8, 9, 10, 11, 12, 13, 14, 15);
    }
    // --- A fragments: element i <-> K = hh*16 + i, M = lh ---
#pragma unroll
    for (int mt = 0; mt < 4; ++mt) {
      int m = wm * 64 + mt * 16 + lh;
      v8bf alo = *(const v8bf*)&sA[cur][m * BK + hh * 16];
      v8bf ahi = *(const v8bf*)&sA[cur][m * BK + hh * 16 + 8];
      v16bf afrag = __builtin_shufflevector(alo, ahi, 0, 1, 2, 3, 4, 5, 6, 7,
                                            8, 9, 10, 11, 12, 13, 14, 15);
#pragma unroll
      for (int nt = 0; nt < 2; ++nt) {
        acc[mt][nt] = __builtin_amdgcn_wmma_f32_16x16x32_bf16(
            false, afrag, false, bfrag[nt], (short)0, acc[mt][nt], false, false);
      }
    }
    async_wait_all();
    __syncthreads();
  }

  // --- epilogue: C/D layout elem i <-> M = i + hh*8, N = lh ---
#pragma unroll
  for (int mt = 0; mt < 4; ++mt) {
#pragma unroll
    for (int nt = 0; nt < 2; ++nt) {
      int n = n0 + wn * 32 + nt * 16 + lh;
#pragma unroll
      for (int i = 0; i < 8; ++i) {
        int m = m0 + wm * 64 + mt * 16 + hh * 8 + i;
        float v = acc[mt][nt][i];
        if (MODE == 0) {
          hidden[((size_t)e * CAP + m) * F + n] = (__bf16)gelu_tanh(v);
        } else {
          int t = s2t[e * CAP + m];
          if (t >= 0) out[(size_t)t * D + n] = v * sgate[e * CAP + m];
        }
      }
    }
  }
}

// ---------- host launcher ----------
extern "C" void kernel_launch(void* const* d_in, const int* in_sizes, int n_in,
                              void* d_out, int out_size, void* d_ws, size_t ws_size,
                              hipStream_t stream) {
  (void)in_sizes; (void)n_in; (void)out_size; (void)ws_size;
  const float* x  = (const float*)d_in[0];   // [B,S,D]
  const float* rw = (const float*)d_in[1];   // [D,E]
  const float* w1 = (const float*)d_in[2];   // [E,D,F]
  const float* w2 = (const float*)d_in[3];   // [E,F,D]
  float* out = (float*)d_out;                // [B,S,D]

  char* ws = (char*)d_ws;
  const size_t nXe  = (size_t)T * D;           // bf16
  const size_t nW1  = (size_t)E * D * F;       // bf16
  const size_t nW2  = (size_t)E * F * D;       // bf16
  const size_t nHid = (size_t)E * CAP * F;     // bf16
  size_t off = 0;
  __bf16* Xe   = (__bf16*)(ws + off); off += nXe  * 2;
  __bf16* W1b  = (__bf16*)(ws + off); off += nW1  * 2;
  __bf16* W2b  = (__bf16*)(ws + off); off += nW2  * 2;
  __bf16* Hid  = (__bf16*)(ws + off); off += nHid * 2;
  int*    eidx = (int*)   (ws + off); off += (size_t)T * 4;
  float*  gate = (float*) (ws + off); off += (size_t)T * 4;
  int*    s2t  = (int*)   (ws + off); off += (size_t)E * CAP * 4;
  float*  sg   = (float*) (ws + off); off += (size_t)E * CAP * 4;

  // 1) zero output (dropped tokens stay 0)
  zero_f32_kernel<<<4096, 256, 0, stream>>>((float4*)out, (int)((size_t)T * D / 4));

  // 2) router
  router_kernel<<<T / 8, 256, 0, stream>>>(x, rw, eidx, gate);

  // 3) capacity positions (sequential per expert, matches reference cumsum order)
  positions_kernel<<<1, 8, 0, stream>>>(eidx, gate, s2t, sg);

  // 4) dispatch + weight conversion to bf16
  dispatch_kernel<<<E * CAP, 256, 0, stream>>>(x, s2t, Xe);
  cvt_bf16_kernel<<<8192, 256, 0, stream>>>((const float4*)w1, (v4bf*)W1b, (int)(nW1 / 4));
  cvt_bf16_kernel<<<8192, 256, 0, stream>>>((const float4*)w2, (v4bf*)W2b, (int)(nW2 / 4));

  // 5) grouped FFN: hidden = gelu(Xe @ W1)   [E,CAP,F]
  {
    dim3 grid(F / BN, CAP / BM, E);
    moe_gemm_kernel<0><<<grid, 256, 0, stream>>>(
        Xe, W1b, (size_t)CAP * D, (size_t)D * F, /*K=*/D, /*lda=*/D, /*ldb=*/F,
        Hid, nullptr, nullptr, nullptr);
  }
  // 6) out[t] = (hidden @ W2)[e,c,:] * gate, scattered to tokens
  {
    dim3 grid(D / BN, CAP / BM, E);
    moe_gemm_kernel<1><<<grid, 256, 0, stream>>>(
        Hid, W2b, (size_t)CAP * F, (size_t)F * D, /*K=*/F, /*lda=*/F, /*ldb=*/D,
        nullptr, out, s2t, sg);
  }
}
